// SparsePooling_16458314678521
// MI455X (gfx1250) — compile-verified, matrix operand order checked by reading, not run on hardware
//
#include <hip/hip_runtime.h>
#include <hip/hip_bf16.h>

typedef __attribute__((ext_vector_type(16))) __bf16 v16bf;
typedef __attribute__((ext_vector_type(8)))  float  v8f;
typedef unsigned int v4u  __attribute__((ext_vector_type(4)));
typedef int          v8i_ __attribute__((ext_vector_type(8)));
typedef int          v4i_ __attribute__((ext_vector_type(4)));

#define B_  8192
#define L_  512
#define H_  1024
#define O_  96
#define E_  8

#define MT  32      // batch rows per block
#define HC  64      // hidden chunk width
#define NT  128     // threads per block = 4 waves (wave32)

// ---------------- prep kernels ----------------

__global__ void cvt_x_kernel(const float* __restrict__ x, __bf16* __restrict__ xb, int n) {
    int i = blockIdx.x * blockDim.x + threadIdx.x;
    if (i < n) xb[i] = (__bf16)x[i];
}

// W1 [E][L][H] f32  ->  w1b [E][H][L] bf16 (transposed: WMMA B-frag wants K contiguous)
__global__ void cvt_w1_kernel(const float* __restrict__ W1, __bf16* __restrict__ w1b) {
    int i = blockIdx.x * blockDim.x + threadIdx.x;
    if (i >= E_ * L_ * H_) return;
    int e = i / (L_ * H_);
    int r = i % (L_ * H_);
    int l = r / H_;
    int h = r % H_;
    w1b[((size_t)(e * H_ + h)) * L_ + l] = (__bf16)W1[i];
}

// W2 [E][H][O] f32  ->  w2b [E][O][H] bf16
__global__ void cvt_w2_kernel(const float* __restrict__ W2, __bf16* __restrict__ w2b) {
    int i = blockIdx.x * blockDim.x + threadIdx.x;
    if (i >= E_ * H_ * O_) return;
    int e = i / (H_ * O_);
    int r = i % (H_ * O_);
    int h = r / O_;
    int o = r % O_;
    w2b[((size_t)(e * O_ + o)) * H_ + h] = (__bf16)W2[i];
}

__global__ void zero_out_kernel(float* __restrict__ out, int n) {
    int i = blockIdx.x * blockDim.x + threadIdx.x;
    if (i < n) out[i] = 0.0f;
}

// ---------------- gating: logits -> top2 -> softmax -> dense [B,E] weights ----------------

__global__ void gate_kernel(const float* __restrict__ x, const float* __restrict__ Wg,
                            const float* __restrict__ bg, float* __restrict__ wgt) {
    int b = blockIdx.x * blockDim.x + threadIdx.x;
    if (b >= B_) return;
    float acc[E_];
#pragma unroll
    for (int e = 0; e < E_; ++e) acc[e] = bg[e];
    const float* xr = x + (size_t)b * L_;
    for (int l = 0; l < L_; ++l) {
        float xv = xr[l];
        const float* wr = Wg + l * E_;
#pragma unroll
        for (int e = 0; e < E_; ++e) acc[e] = fmaf(xv, wr[e], acc[e]);
    }
    int i1 = 0; float v1 = acc[0];
#pragma unroll
    for (int e = 1; e < E_; ++e) if (acc[e] > v1) { v1 = acc[e]; i1 = e; }
    int i2 = -1; float v2 = -3.4e38f;
#pragma unroll
    for (int e = 0; e < E_; ++e) if (e != i1 && acc[e] > v2) { v2 = acc[e]; i2 = e; }
    float p2 = __expf(v2 - v1);
    float inv = 1.0f / (1.0f + p2);
    float* wr = wgt + (size_t)b * E_;
#pragma unroll
    for (int e = 0; e < E_; ++e) wr[e] = 0.0f;
    wr[i1] = inv;
    wr[i2] = p2 * inv;
}

// ---------------- fused MoE expert kernel ----------------
__global__ __launch_bounds__(NT)
void moe_kernel(const __bf16* __restrict__ xb, const __bf16* __restrict__ w1b,
                const __bf16* __restrict__ w2b, const float* __restrict__ b1,
                const float* __restrict__ b2, const float* __restrict__ wgt,
                float* __restrict__ out) {
    __shared__ __align__(32) __bf16 xs[MT * L_];   // 32 KB
    __shared__ __align__(32) __bf16 hs[MT * HC];   // 4 KB

    const int e   = blockIdx.x % E_;
    const int m0  = (blockIdx.x / E_) * MT;
    const int tid = threadIdx.x;
    const int wave  = tid >> 5;
    const int lane  = tid & 31;
    const int lhalf = lane >> 4;   // 0: lanes 0-15, 1: lanes 16-31
    const int lmod  = lane & 15;

    // ---- stage x tile: one contiguous 32 KB region -> LDS via Tensor Data Mover ----
#if __has_builtin(__builtin_amdgcn_tensor_load_to_lds)
    if (wave == 0) {
        const unsigned TILE_ELEMS = MT * L_;                 // 16384 bf16 = 32 KB
        const unsigned lds_base   = (unsigned)(size_t)(void*)xs;
        const unsigned long long ga =
            (unsigned long long)(const void*)(xb + (size_t)m0 * L_);
        // D# group 0: [1:0]=count=1, [63:32]=lds_addr, [120:64]=global_addr, [127:126]=type=2
        v4u g0 = { 1u,
                   lds_base,
                   (unsigned)(ga & 0xFFFFFFFFu),
                   (unsigned)((ga >> 32) & 0x01FFFFFFu) | (2u << 30) };
        // D# group 1: [17:16]=data_size(1 -> 2B), tensor_dim0 @bits[79:48],
        //             tile_dim0 @bits[127:112]; 1-D tile (tile_dim1/2 = 0), strides unused
        v8i_ g1 = { (int)0x00010000u,                          // wg_mask=0, data_size=2B
                    (int)((TILE_ELEMS & 0xFFFFu) << 16),       // tensor_dim0[15:0]
                    (int)(TILE_ELEMS >> 16),                   // tensor_dim0[31:16]
                    (int)((TILE_ELEMS & 0xFFFFu) << 16),       // tile_dim0 in high 16
                    0, 0, 0, 0 };
        v4i_ g2 = { 0, 0, 0, 0 };
        v4i_ g3 = { 0, 0, 0, 0 };
        v8i_ g4 = { 0, 0, 0, 0, 0, 0, 0, 0 };
        __builtin_amdgcn_tensor_load_to_lds(g0, g1, g2, g3, g4, 0);
#if __has_builtin(__builtin_amdgcn_s_wait_tensorcnt)
        __builtin_amdgcn_s_wait_tensorcnt(0);
#else
        asm volatile("s_wait_tensorcnt 0x0" ::: "memory");
#endif
    }
#else
    {
        const uint4* gx = (const uint4*)(xb + (size_t)m0 * L_);
        uint4* sx = (uint4*)xs;
        const int total = (MT * L_) / 8;
#pragma unroll
        for (int it = 0; it < total / NT; ++it)
            sx[it * NT + tid] = gx[it * NT + tid];
    }
#endif
    __syncthreads();

    // GEMM1 tile mapping: (2 mtiles) x (4 ntiles); wave -> 2 frags
    const int mt1 = wave & 1;
    const int ntp = wave >> 1;
    // GEMM2 tile mapping: (2 mtiles) x (6 otiles); wave -> 3 frags
    const int mt2 = wave & 1;
    const int op  = wave >> 1;

    v8f oacc[3];
#pragma unroll
    for (int j = 0; j < 3; ++j)
#pragma unroll
        for (int v = 0; v < 8; ++v) oacc[j][v] = 0.0f;

    for (int hc = 0; hc < H_; hc += HC) {
        __syncthreads();   // everyone done reading hs from the previous chunk

        // ---- GEMM1: hidden[MT][HC] = x_tile[MT][L] @ W1_e[:, hc:hc+HC] ----
        v8f c1[2];
#pragma unroll
        for (int j = 0; j < 2; ++j)
#pragma unroll
            for (int v = 0; v < 8; ++v) c1[j][v] = 0.0f;

        for (int kk = 0; kk < L_; kk += 32) {
            const v16bf a = *(const v16bf*)&xs[(mt1 * 16 + lmod) * L_ + kk + lhalf * 16];
#pragma unroll
            for (int j = 0; j < 2; ++j) {
                const int n = hc + (ntp * 2 + j) * 16 + lmod;
                const v16bf bm = *(const v16bf*)&w1b[((size_t)(e * H_ + n)) * L_ + kk + lhalf * 16];
                c1[j] = __builtin_amdgcn_wmma_f32_16x16x32_bf16(
                            false, a, false, bm, (short)0, c1[j], false, false);
            }
        }

        // speculative prefetch of next W1 chunk rows (global_prefetch_b8)
        if (hc + HC < H_)
            __builtin_prefetch(&w1b[((size_t)(e * H_ + hc + HC + ntp * 32 + lmod)) * L_], 0, 0);

        // ---- epilogue: +b1, relu, cast bf16, write hidden chunk to LDS ----
#pragma unroll
        for (int j = 0; j < 2; ++j) {
            const int nloc = (ntp * 2 + j) * 16 + lmod;
            const float bias = b1[e * H_ + hc + nloc];
#pragma unroll
            for (int v = 0; v < 8; ++v) {
                float val = c1[j][v] + bias;
                val = val > 0.0f ? val : 0.0f;
                hs[(mt1 * 16 + v + lhalf * 8) * HC + nloc] = (__bf16)val;
            }
        }
        __syncthreads();

        // ---- GEMM2: out_acc += hidden[MT][HC] @ W2_e[hc:hc+HC, :] ----
#pragma unroll
        for (int kk = 0; kk < HC; kk += 32) {
            const v16bf a2 = *(const v16bf*)&hs[(mt2 * 16 + lmod) * HC + kk + lhalf * 16];
#pragma unroll
            for (int j2 = 0; j2 < 3; ++j2) {
                const int o = (op * 3 + j2) * 16 + lmod;
                const v16bf bm = *(const v16bf*)&w2b[((size_t)(e * O_ + o)) * H_ + hc + kk + lhalf * 16];
                oacc[j2] = __builtin_amdgcn_wmma_f32_16x16x32_bf16(
                               false, a2, false, bm, (short)0, oacc[j2], false, false);
            }
        }
    }

    // ---- epilogue: (acc + b2) * gate_weight -> atomicAdd into out ----
#pragma unroll
    for (int j2 = 0; j2 < 3; ++j2) {
        const int col = (op * 3 + j2) * 16 + lmod;
        const float bias2 = b2[e * O_ + col];
#pragma unroll
        for (int v = 0; v < 8; ++v) {
            const int gr = m0 + mt2 * 16 + v + lhalf * 8;
            const float w = wgt[(size_t)gr * E_ + e];
            atomicAdd(&out[(size_t)gr * O_ + col], w * (oacc[j2][v] + bias2));
        }
    }
}

// ---------------- launch ----------------

extern "C" void kernel_launch(void* const* d_in, const int* in_sizes, int n_in,
                              void* d_out, int out_size, void* d_ws, size_t ws_size,
                              hipStream_t stream) {
    const float* x  = (const float*)d_in[0];
    // d_in[1] = insample_mask (all-ones; unused by the reference math)
    const float* W1 = (const float*)d_in[2];
    const float* b1 = (const float*)d_in[3];
    const float* W2 = (const float*)d_in[4];
    const float* b2 = (const float*)d_in[5];
    const float* Wg = (const float*)d_in[6];
    const float* bg = (const float*)d_in[7];
    float* out = (float*)d_out;

    const size_t XB_BYTES  = (size_t)B_ * L_ * 2;
    const size_t W1B_BYTES = (size_t)E_ * H_ * L_ * 2;
    const size_t W2B_BYTES = (size_t)E_ * O_ * H_ * 2;
    char* ws = (char*)d_ws;
    __bf16* xb  = (__bf16*)ws;
    __bf16* w1b = (__bf16*)(ws + XB_BYTES);
    __bf16* w2b = (__bf16*)(ws + XB_BYTES + W1B_BYTES);
    float*  wgt = (float*) (ws + XB_BYTES + W1B_BYTES + W2B_BYTES);

    cvt_x_kernel <<<(B_ * L_ + 255) / 256, 256, 0, stream>>>(x, xb, B_ * L_);
    cvt_w1_kernel<<<(E_ * L_ * H_ + 255) / 256, 256, 0, stream>>>(W1, w1b);
    cvt_w2_kernel<<<(E_ * H_ * O_ + 255) / 256, 256, 0, stream>>>(W2, w2b);
    gate_kernel  <<<(B_ + 255) / 256, 256, 0, stream>>>(x, Wg, bg, wgt);
    zero_out_kernel<<<(B_ * O_ + 255) / 256, 256, 0, stream>>>(out, B_ * O_);

    moe_kernel<<<dim3((B_ / MT) * E_), NT, 0, stream>>>(xb, w1b, w2b, b1, b2, wgt, out);
}
